// FastTransformerDecoder_57767310131321
// MI455X (gfx1250) — matrix operand order repaired and checked by reading, not compiled, not run on hardware
//
#include <hip/hip_runtime.h>

// ---------------- problem constants ----------------
#define N_LAYER  6
#define N_HEAD   16
#define DMODEL   1024
#define DFF      4096
#define DHEAD    64
#define MFEAT    128
#define CHUNK_S  128
#define BB       4
#define SSEQ     2048
#define NCHUNK   (SSEQ / CHUNK_S)     // 16
#define LN_EPS   1e-5f
#define ATT_EPS  1e-6f

typedef __bf16 bf16_t;
typedef bf16_t v16bf __attribute__((ext_vector_type(16)));
typedef bf16_t v8bf  __attribute__((ext_vector_type(8)));
typedef bf16_t v4bf  __attribute__((ext_vector_type(4)));
typedef float  v8f   __attribute__((ext_vector_type(8)));

// ---- optional CDNA5 async global->LDS copy path (ASYNCcnt) ----
#if defined(__HIP_DEVICE_COMPILE__) && defined(__has_builtin)
#if __has_builtin(__builtin_amdgcn_global_load_async_to_lds_b128) && \
    __has_builtin(__builtin_amdgcn_s_wait_asynccnt)
#define HAVE_ASYNC_LDS 1
#endif
#endif

#ifdef HAVE_ASYNC_LDS
typedef int v4i_async __attribute__((vector_size(4 * sizeof(int))));
typedef __attribute__((address_space(1))) v4i_async* as1_v4i;
typedef __attribute__((address_space(3))) v4i_async* as3_v4i;
__device__ __forceinline__ void async_cp16(const void* g, void* l) {
  v4i_async* gp = reinterpret_cast<v4i_async*>(const_cast<void*>(g));
  v4i_async* lp = reinterpret_cast<v4i_async*>(l);
  __builtin_amdgcn_global_load_async_to_lds_b128((as1_v4i)gp, (as3_v4i)lp, 0, 0);
}
#endif

__device__ __forceinline__ v8f zero8() {
  v8f z;
#pragma unroll
  for (int i = 0; i < 8; ++i) z[i] = 0.f;
  return z;
}

// pack 8 f32 -> 8 bf16 and store as one b128
__device__ __forceinline__ void store8(bf16_t* dst, float4 a, float4 b) {
  __align__(16) bf16_t t[8];
  t[0] = (bf16_t)a.x; t[1] = (bf16_t)a.y; t[2] = (bf16_t)a.z; t[3] = (bf16_t)a.w;
  t[4] = (bf16_t)b.x; t[5] = (bf16_t)b.y; t[6] = (bf16_t)b.z; t[7] = (bf16_t)b.w;
  *(v8bf*)dst = *(const v8bf*)t;
}

// A-fragment (16x32 bf16, MxK): lane->M=lane&15, K = kb..kb+7, kb+16..kb+23, kb=(lane>>4)*8
__device__ __forceinline__ v16bf frag_a(const bf16_t* __restrict__ tile, int ld) {
  int lane = threadIdx.x & 31;
  const bf16_t* p = tile + (lane & 15) * ld + ((lane >> 4) << 3);
  v8bf lo = *(const v8bf*)p;
  v8bf hi = *(const v8bf*)(p + 16);
  v16bf r;
#pragma unroll
  for (int i = 0; i < 8; ++i) { r[i] = lo[i]; r[i + 8] = hi[i]; }
  return r;
}

// B-fragment (32x16 bf16, KxN) from k-contiguous transposed tile Bt[n][k]
__device__ __forceinline__ v16bf frag_b(const bf16_t* __restrict__ tile, int ld) {
  int lane = threadIdx.x & 31;
  const bf16_t* p = tile + (lane & 15) * ld + ((lane >> 4) << 3);
  v8bf lo = *(const v8bf*)p;
  v8bf hi = *(const v8bf*)(p + 16);
  v16bf r;
#pragma unroll
  for (int i = 0; i < 8; ++i) { r[i] = lo[i]; r[i + 8] = hi[i]; }
  return r;
}

#define WMMA_BF16(a, b, c) \
  __builtin_amdgcn_wmma_f32_16x16x32_bf16(false, (a), false, (b), (short)0, (c), false, false)

// =====================================================================
// one-time weight transpose+convert: Wt[N][K] bf16  <-  W[K][N] f32
// =====================================================================
__global__ __launch_bounds__(256) void k_prep_wT(
    const float* __restrict__ W, bf16_t* __restrict__ Wt, int K, int N)
{
  __shared__ float T[32][33];
  const int tid = threadIdx.x;
  const int n0 = blockIdx.x * 32, k0 = blockIdx.y * 32;
  {
    int r = tid >> 3, c = (tid & 7) << 2;
    float4 v = *(const float4*)(W + (size_t)(k0 + r) * N + n0 + c);
    T[c + 0][r] = v.x; T[c + 1][r] = v.y; T[c + 2][r] = v.z; T[c + 3][r] = v.w;
  }
  __syncthreads();
  {
    int n = tid >> 3, kc = (tid & 7) << 2;
    __align__(8) bf16_t t[4];
    t[0] = (bf16_t)T[n][kc + 0]; t[1] = (bf16_t)T[n][kc + 1];
    t[2] = (bf16_t)T[n][kc + 2]; t[3] = (bf16_t)T[n][kc + 3];
    *(v4bf*)(Wt + (size_t)(n0 + n) * K + k0 + kc) = *(const v4bf*)t;
  }
}

// =====================================================================
// GEMM: C[M,N] = act(A[M,K] f32 x W^T (Wt[N][K] bf16) + bias[N])
// block 256 threads = 8 waves (2x4), tile 128(M) x 256(N), K-step 32.
// Each wave: 64x64 -> 16 WMMAs per k-step.
// =====================================================================
__global__ __launch_bounds__(256) void k_gemm_bias(
    const float* __restrict__ A, const bf16_t* __restrict__ Wt,
    const float* __restrict__ bias, float* __restrict__ C,
    int M, int N, int K, int relu)
{
  __shared__ __align__(16) bf16_t Alds[128][40];
  __shared__ __align__(16) bf16_t Blds[256][40];   // Bt[n][k], k-contiguous
  const int tid  = threadIdx.x;
  const int wave = tid >> 5, lane = tid & 31;
  const int wm = (wave >> 2) * 64;     // 0 / 64
  const int wn = (wave & 3) * 64;      // 0..192
  const int bm = blockIdx.y * 128, bn = blockIdx.x * 256;

  const int arow = tid >> 1, acb = (tid & 1) << 4;
  const float*  gA = A  + (size_t)(bm + arow) * K + acb;
  const bf16_t* gB = Wt + (size_t)(bn + tid) * K;

  v8f acc[4][4];
#pragma unroll
  for (int i = 0; i < 4; ++i)
#pragma unroll
    for (int j = 0; j < 4; ++j) acc[i][j] = zero8();

  for (int k0 = 0; k0 < K; k0 += 32) {
    if (k0 + 32 < K) {   // prefetch next k-tile (global_prefetch_b8)
      __builtin_prefetch(gA + k0 + 32, 0, 1);
      __builtin_prefetch(gB + k0 + 32, 0, 1);
    }
    {   // stage A: 128x32 f32 -> bf16, packed b128 stores
      const float* g = gA + k0;
      float4 v0 = *(const float4*)(g + 0),  v1 = *(const float4*)(g + 4);
      float4 v2 = *(const float4*)(g + 8),  v3 = *(const float4*)(g + 12);
      store8(&Alds[arow][acb + 0], v0, v1);
      store8(&Alds[arow][acb + 8], v2, v3);
    }
    {   // stage B: row of bf16 weights, pure byte copy (64B per thread)
      const bf16_t* g = gB + k0;
      bf16_t* d = &Blds[tid][0];
#ifdef HAVE_ASYNC_LDS
#pragma unroll
      for (int i = 0; i < 4; ++i) async_cp16(g + i * 8, d + i * 8);
#else
#pragma unroll
      for (int i = 0; i < 4; ++i) *(v8bf*)(d + i * 8) = *(const v8bf*)(g + i * 8);
#endif
    }
#ifdef HAVE_ASYNC_LDS
    __builtin_amdgcn_s_wait_asynccnt(0);
#endif
    __syncthreads();
    v16bf bfr[4];
#pragma unroll
    for (int nt = 0; nt < 4; ++nt) bfr[nt] = frag_b(&Blds[wn + nt * 16][0], 40);
#pragma unroll
    for (int mt = 0; mt < 4; ++mt) {
      v16bf af = frag_a(&Alds[wm + mt * 16][0], 40);
#pragma unroll
      for (int nt = 0; nt < 4; ++nt)
        acc[mt][nt] = WMMA_BF16(af, bfr[nt], acc[mt][nt]);
    }
    __syncthreads();
  }

  const int colL = lane & 15, rowoff = (lane >> 4) << 3;
#pragma unroll
  for (int mt = 0; mt < 4; ++mt)
#pragma unroll
    for (int nt = 0; nt < 4; ++nt) {
      int gc = bn + wn + nt * 16 + colL;
      float bv = bias[gc];
#pragma unroll
      for (int r = 0; r < 8; ++r) {
        int gr = bm + wm + mt * 16 + rowoff + r;
        float vv = acc[mt][nt][r] + bv;
        if (relu) vv = fmaxf(vv, 0.f);
        C[(size_t)gr * N + gc] = vv;
      }
    }
}

// =====================================================================
// FAVOR phi: u = (x*Dh^-0.25) x omega (128x64x64 WMMA),
// phi = [exp(u-nrm), exp(-u-nrm)] * M^-0.5 -> P[bh*S+s][128]
// =====================================================================
__global__ __launch_bounds__(256) void k_phi(
    const float* __restrict__ X, const float* __restrict__ omega,
    float* __restrict__ P)
{
  __shared__ __align__(16) bf16_t Xl[128][72];
  __shared__ __align__(16) bf16_t Ot[64][72];
  __shared__ float part[256];
  __shared__ float nrm[128];
  const float xscale = 0.35355339059327379f;   // 64^-0.25
  const float pscale = 0.08838834764831845f;   // 128^-0.5
  const int tid = threadIdx.x;
  const int sc = blockIdx.x, bh = blockIdx.y;
  const int b = bh >> 4, h = bh & 15;

  {   // stage x (scaled) + partial row sumsq, packed stores
    int row = tid >> 1, cb = (tid & 1) << 5;
    const float* g = X + ((size_t)(b * SSEQ + sc * 128 + row)) * DMODEL + h * 64 + cb;
    float ss = 0.f;
#pragma unroll
    for (int i = 0; i < 32; i += 8) {
      float4 v0 = *(const float4*)(g + i), v1 = *(const float4*)(g + i + 4);
      float xs[8] = {v0.x, v0.y, v0.z, v0.w, v1.x, v1.y, v1.z, v1.w};
      __align__(16) bf16_t t[8];
#pragma unroll
      for (int e = 0; e < 8; ++e) { float x = xs[e] * xscale; ss += x * x; t[e] = (bf16_t)x; }
      *(v8bf*)&Xl[row][cb + i] = *(const v8bf*)t;
    }
    part[tid] = 0.5f * ss;
  }
#pragma unroll
  for (int i = 0; i < 16; ++i) {   // omega transposed (tiny)
    int idx = tid * 16 + i;
    int k = idx >> 6, n = idx & 63;
    Ot[n][k] = (bf16_t)omega[k * 64 + n];
  }
  __syncthreads();
  if (tid < 128) nrm[tid] = part[2 * tid] + part[2 * tid + 1];
  __syncthreads();

  const int wave = tid >> 5, lane = tid & 31;
  const int wm = (wave >> 2) * 64, wn = (wave & 3) * 16;
  v8f acc[4];
#pragma unroll
  for (int i = 0; i < 4; ++i) acc[i] = zero8();
#pragma unroll
  for (int k0 = 0; k0 < 64; k0 += 32) {
    v16bf bf = frag_b(&Ot[wn][k0], 72);
#pragma unroll
    for (int mt = 0; mt < 4; ++mt) {
      v16bf af = frag_a(&Xl[wm + mt * 16][k0], 72);
      acc[mt] = WMMA_BF16(af, bf, acc[mt]);
    }
  }
  const int colL = lane & 15, rowoff = (lane >> 4) << 3;
  const size_t obase = (size_t)bh * SSEQ + sc * 128;
#pragma unroll
  for (int mt = 0; mt < 4; ++mt)
#pragma unroll
    for (int r = 0; r < 8; ++r) {
      int row = wm + mt * 16 + rowoff + r;
      float u = acc[mt][r], nr = nrm[row];
      float* o = P + (obase + row) * MFEAT;
      o[wn + colL]      = __expf( u - nr) * pscale;
      o[wn + colL + 64] = __expf(-u - nr) * pscale;
    }
}

// =====================================================================
// per-chunk kv = K^T V (128x64x128 WMMA) + ksum
// =====================================================================
__global__ __launch_bounds__(256) void k_chunk_kv(
    const float* __restrict__ Kp, const float* __restrict__ V,
    float* __restrict__ KV, float* __restrict__ KS)
{
  __shared__ __align__(16) bf16_t Kt[128][136];   // [m][l]
  __shared__ __align__(16) bf16_t Vt[64][136];    // [d][l]
  const int tid = threadIdx.x;
  const int c = blockIdx.x, bh = blockIdx.y;
  const int b = bh >> 4, h = bh & 15;
  {
    int l = tid >> 1, mb = (tid & 1) << 6;
    const float* g = Kp + ((size_t)bh * SSEQ + c * 128 + l) * MFEAT + mb;
#pragma unroll
    for (int i = 0; i < 64; i += 4) {
      float4 v = *(const float4*)(g + i);
      Kt[mb + i + 0][l] = (bf16_t)v.x; Kt[mb + i + 1][l] = (bf16_t)v.y;
      Kt[mb + i + 2][l] = (bf16_t)v.z; Kt[mb + i + 3][l] = (bf16_t)v.w;
    }
  }
  {
    int l = tid >> 1, db = (tid & 1) << 5;
    const float* g = V + ((size_t)b * SSEQ + c * 128 + l) * DMODEL + h * 64 + db;
#pragma unroll
    for (int i = 0; i < 32; i += 4) {
      float4 v = *(const float4*)(g + i);
      Vt[db + i + 0][l] = (bf16_t)v.x; Vt[db + i + 1][l] = (bf16_t)v.y;
      Vt[db + i + 2][l] = (bf16_t)v.z; Vt[db + i + 3][l] = (bf16_t)v.w;
    }
  }
  __syncthreads();
  if (tid < 128) {
    float s = 0.f;
#pragma unroll 8
    for (int l = 0; l < 128; ++l) s += (float)Kt[tid][l];
    KS[((size_t)bh * NCHUNK + c) * 128 + tid] = s;
  }
  const int wave = tid >> 5, lane = tid & 31;
  const int wm = (wave >> 2) * 64, wn = (wave & 3) * 16;
  v8f acc[4];
#pragma unroll
  for (int i = 0; i < 4; ++i) acc[i] = zero8();
#pragma unroll
  for (int k0 = 0; k0 < 128; k0 += 32) {
    v16bf bf = frag_b(&Vt[wn][k0], 136);
#pragma unroll
    for (int mt = 0; mt < 4; ++mt) {
      v16bf af = frag_a(&Kt[wm + mt * 16][k0], 136);
      acc[mt] = WMMA_BF16(af, bf, acc[mt]);
    }
  }
  const int colL = lane & 15, rowoff = (lane >> 4) << 3;
  float* out = KV + ((size_t)bh * NCHUNK + c) * (128 * 64);
#pragma unroll
  for (int mt = 0; mt < 4; ++mt)
#pragma unroll
    for (int r = 0; r < 8; ++r) {
      int m = wm + mt * 16 + rowoff + r;
      out[m * 64 + wn + colL] = acc[mt][r];
    }
}

// exclusive prefix over chunks, in place; one block per (b,h)
__global__ __launch_bounds__(256) void k_scan(float* __restrict__ KV, float* __restrict__ KS)
{
  const int bh = blockIdx.x, tid = threadIdx.x;
  float* kvb = KV + (size_t)bh * NCHUNK * 128 * 64;
  for (int p = tid; p < 128 * 64; p += 256) {
    float run = 0.f;
#pragma unroll
    for (int c = 0; c < NCHUNK; ++c) {
      float t = kvb[(size_t)c * 128 * 64 + p];
      kvb[(size_t)c * 128 * 64 + p] = run;
      run += t;
    }
  }
  if (tid < 128) {
    float* ksb = KS + (size_t)bh * NCHUNK * 128;
    float run = 0.f;
#pragma unroll
    for (int c = 0; c < NCHUNK; ++c) {
      float t = ksb[c * 128 + tid];
      ksb[c * 128 + tid] = run;
      run += t;
    }
  }
}

// =====================================================================
// attention output per chunk:
//   S = tril(Q K^T); den = rowsum(S) + Q.ks_pre; num = S V + Q kv_pre
//   out = num / (den + eps) scattered to [b,s,h*64+d]
// =====================================================================
__global__ __launch_bounds__(256) void k_attn(
    const float* __restrict__ Qp, const float* __restrict__ Kp,
    const float* __restrict__ V,  const float* __restrict__ KVpre,
    const float* __restrict__ KSpre, float* __restrict__ Aout)
{
  __shared__ __align__(16) bf16_t Qs[128][136];     // [i][m]
  __shared__ __align__(16) bf16_t Bufa[128][136];   // K[j][m], then masked S[i][j]
  __shared__ __align__(16) bf16_t Vt[64][136];      // [d][j]
  __shared__ __align__(16) bf16_t KVt[64][136];     // [d][m]
  __shared__ float den[128];
  const int tid = threadIdx.x;
  const int c = blockIdx.x, bh = blockIdx.y;
  const int b = bh >> 4, h = bh & 15;
  {
    int i = tid >> 1, mb = (tid & 1) << 6;
    const float* gq = Qp + ((size_t)bh * SSEQ + c * 128 + i) * MFEAT + mb;
    const float* gk = Kp + ((size_t)bh * SSEQ + c * 128 + i) * MFEAT + mb;
#pragma unroll
    for (int j = 0; j < 64; j += 8) {
      float4 a0 = *(const float4*)(gq + j), a1 = *(const float4*)(gq + j + 4);
      store8(&Qs[i][mb + j], a0, a1);
      float4 b0 = *(const float4*)(gk + j), b1 = *(const float4*)(gk + j + 4);
      store8(&Bufa[i][mb + j], b0, b1);
    }
  }
  {
    int l = tid >> 1, db = (tid & 1) << 5;
    const float* g = V + ((size_t)b * SSEQ + c * 128 + l) * DMODEL + h * 64 + db;
#pragma unroll
    for (int i = 0; i < 32; i += 4) {
      float4 v = *(const float4*)(g + i);
      Vt[db + i + 0][l] = (bf16_t)v.x; Vt[db + i + 1][l] = (bf16_t)v.y;
      Vt[db + i + 2][l] = (bf16_t)v.z; Vt[db + i + 3][l] = (bf16_t)v.w;
    }
  }
  {
    int m = tid >> 1, db = (tid & 1) << 5;
    const float* g = KVpre + ((size_t)bh * NCHUNK + c) * (128 * 64) + (size_t)m * 64 + db;
#pragma unroll
    for (int i = 0; i < 32; i += 4) {
      float4 v = *(const float4*)(g + i);
      KVt[db + i + 0][m] = (bf16_t)v.x; KVt[db + i + 1][m] = (bf16_t)v.y;
      KVt[db + i + 2][m] = (bf16_t)v.z; KVt[db + i + 3][m] = (bf16_t)v.w;
    }
  }
  __syncthreads();

  const int wave = tid >> 5, lane = tid & 31;
  const int colL = lane & 15, rowoff = (lane >> 4) << 3;
  const int wm = (wave >> 2) * 64;
  {   // scores 128x128 over m
    const int wn = (wave & 3) * 32;
    v8f s[4][2];
#pragma unroll
    for (int i = 0; i < 4; ++i) { s[i][0] = zero8(); s[i][1] = zero8(); }
#pragma unroll
    for (int k0 = 0; k0 < 128; k0 += 32) {
      v16bf b0 = frag_b(&Bufa[wn +  0][k0], 136);
      v16bf b1 = frag_b(&Bufa[wn + 16][k0], 136);
#pragma unroll
      for (int mt = 0; mt < 4; ++mt) {
        v16bf af = frag_a(&Qs[wm + mt * 16][k0], 136);
        s[mt][0] = WMMA_BF16(af, b0, s[mt][0]);
        s[mt][1] = WMMA_BF16(af, b1, s[mt][1]);
      }
    }
    __syncthreads();   // done reading K from Bufa
#pragma unroll
    for (int mt = 0; mt < 4; ++mt)
#pragma unroll
      for (int nt = 0; nt < 2; ++nt)
#pragma unroll
        for (int r = 0; r < 8; ++r) {
          int i = wm + mt * 16 + rowoff + r;
          int j = wn + nt * 16 + colL;
          float v = (j <= i) ? s[mt][nt][r] : 0.f;
          Bufa[i][j] = (bf16_t)v;   // masked scores
        }
  }
  __syncthreads();
  if (tid < 128) {
    float d1 = 0.f;
#pragma unroll 8
    for (int j = 0; j < 128; ++j) d1 += (float)Bufa[tid][j];
    const float* ks = KSpre + ((size_t)bh * NCHUNK + c) * 128;
    float d2 = 0.f;
#pragma unroll 8
    for (int m = 0; m < 128; ++m) d2 += (float)Qs[tid][m] * ks[m];
    den[tid] = d1 + d2 + ATT_EPS;
  }
  __syncthreads();
  {   // num = S*V (over j) + Q*kv_pre (over m)
    const int wn = (wave & 3) * 16;
    v8f acc[4];
#pragma unroll
    for (int i = 0; i < 4; ++i) acc[i] = zero8();
#pragma unroll
    for (int k0 = 0; k0 < 128; k0 += 32) {
      v16bf bv = frag_b(&Vt[wn][k0], 136);
      v16bf bk = frag_b(&KVt[wn][k0], 136);
#pragma unroll
      for (int mt = 0; mt < 4; ++mt) {
        v16bf as = frag_a(&Bufa[wm + mt * 16][k0], 136);
        acc[mt] = WMMA_BF16(as, bv, acc[mt]);
        v16bf aq = frag_a(&Qs[wm + mt * 16][k0], 136);
        acc[mt] = WMMA_BF16(aq, bk, acc[mt]);
      }
    }
#pragma unroll
    for (int mt = 0; mt < 4; ++mt)
#pragma unroll
      for (int r = 0; r < 8; ++r) {
        int i = wm + mt * 16 + rowoff + r;
        float o = acc[mt][r] / den[i];
        Aout[((size_t)b * SSEQ + c * 128 + i) * DMODEL + h * 64 + wn + colL] = o;
      }
  }
}

// residual add + layernorm: O = LN(A + Bx)*g + be ; one row per block
__global__ __launch_bounds__(256) void k_ln(
    const float* __restrict__ A, const float* __restrict__ Bx,
    const float* __restrict__ g, const float* __restrict__ be,
    float* __restrict__ O)
{
  __shared__ float red[256];
  const int row = blockIdx.x, tid = threadIdx.x;
  const float* a  = A  + (size_t)row * DMODEL;
  const float* bb = Bx + (size_t)row * DMODEL;
  float x[4]; float s = 0.f;
#pragma unroll
  for (int i = 0; i < 4; ++i) { int idx = tid + i * 256; x[i] = a[idx] + bb[idx]; s += x[i]; }
  red[tid] = s; __syncthreads();
  for (int st = 128; st > 0; st >>= 1) { if (tid < st) red[tid] += red[tid + st]; __syncthreads(); }
  float mu = red[0] * (1.f / DMODEL);
  __syncthreads();
  float v = 0.f;
#pragma unroll
  for (int i = 0; i < 4; ++i) { float d = x[i] - mu; v += d * d; }
  red[tid] = v; __syncthreads();
  for (int st = 128; st > 0; st >>= 1) { if (tid < st) red[tid] += red[tid + st]; __syncthreads(); }
  float rstd = rsqrtf(red[0] * (1.f / DMODEL) + LN_EPS);
#pragma unroll
  for (int i = 0; i < 4; ++i) {
    int idx = tid + i * 256;
    O[(size_t)row * DMODEL + idx] = (x[i] - mu) * rstd * g[idx] + be[idx];
  }
}

// =====================================================================
extern "C" void kernel_launch(void* const* d_in, const int* in_sizes, int n_in,
                              void* d_out, int out_size, void* d_ws, size_t ws_size,
                              hipStream_t stream) {
  const float* X   = (const float*)d_in[0];
  const float* Wq  = (const float*)d_in[1];
  const float* bq  = (const float*)d_in[2];
  const float* Wk  = (const float*)d_in[3];
  const float* bk  = (const float*)d_in[4];
  const float* Wv  = (const float*)d_in[5];
  const float* bv  = (const float*)d_in[6];
  const float* Wo  = (const float*)d_in[7];
  const float* bo  = (const float*)d_in[8];
  const float* Om  = (const float*)d_in[9];
  const float* g1  = (const float*)d_in[10];
  const float* be1 = (const float*)d_in[11];
  const float* W1  = (const float*)d_in[12];
  const float* bf1 = (const float*)d_in[13];
  const float* W2  = (const float*)d_in[14];
  const float* bf2 = (const float*)d_in[15];
  const float* g2  = (const float*)d_in[16];
  const float* be2 = (const float*)d_in[17];

  const size_t SZ  = (size_t)BB * SSEQ * DMODEL;    // 8M floats
  const size_t MM  = (size_t)DMODEL * DMODEL;       // 1M
  const size_t MF  = (size_t)DMODEL * DFF;          // 4M
  float* ws = (float*)d_ws;
  float* R  = ws;
  float* B1 = ws + 1 * SZ;
  float* B2 = ws + 2 * SZ;
  float* B3 = ws + 3 * SZ;
  float* QP = ws + 4 * SZ;   // 2*SZ
  float* KP = ws + 6 * SZ;   // 2*SZ
  float* KV = ws + 8 * SZ;   // SZ
  float* KS = ws + 9 * SZ;   // 131072
  float* Y1 = ws + 9 * SZ + (1 << 17);   // SZ
  bf16_t* WB  = (bf16_t*)(ws + 10 * SZ + (1 << 17));
  bf16_t* WqT = WB;                       // 6 * 1M bf16 each for q,k,v,o
  bf16_t* WkT = WB + 6 * MM;
  bf16_t* WvT = WB + 12 * MM;
  bf16_t* WoT = WB + 18 * MM;
  bf16_t* W1T = WB + 24 * MM;             // 6 * 4M : [DFF][DMODEL]
  bf16_t* W2T = WB + 24 * MM + 6 * MF;    // 6 * 4M : [DMODEL][DFF]
  const size_t need = (10 * SZ + (size_t)(1 << 17)) * sizeof(float)
                    + (24 * MM + 12 * MF) * sizeof(bf16_t);
  if (ws_size < need) return;

  (void)hipMemcpyAsync(R, X, SZ * sizeof(float), hipMemcpyDeviceToDevice, stream);

  const dim3 blk(256);
  const dim3 gT1(DMODEL / 32, DMODEL / 32);
  const dim3 gTf1(DFF / 32, DMODEL / 32);
  const dim3 gTf2(DMODEL / 32, DFF / 32);
  for (int l = 0; l < N_LAYER; ++l) {   // one-time weight prep (bf16, transposed)
    k_prep_wT<<<gT1,  blk, 0, stream>>>(Wq + l * MM, WqT + l * MM, DMODEL, DMODEL);
    k_prep_wT<<<gT1,  blk, 0, stream>>>(Wk + l * MM, WkT + l * MM, DMODEL, DMODEL);
    k_prep_wT<<<gT1,  blk, 0, stream>>>(Wv + l * MM, WvT + l * MM, DMODEL, DMODEL);
    k_prep_wT<<<gT1,  blk, 0, stream>>>(Wo + l * MM, WoT + l * MM, DMODEL, DMODEL);
    k_prep_wT<<<gTf1, blk, 0, stream>>>(W1 + l * MF, W1T + l * MF, DMODEL, DFF);
    k_prep_wT<<<gTf2, blk, 0, stream>>>(W2 + l * MF, W2T + l * MF, DFF, DMODEL);
  }

  const dim3 gProj(DMODEL / 256, (BB * SSEQ) / 128);   // (4,64)
  const dim3 gAttn(NCHUNK, BB * N_HEAD);               // (16,64)
  const dim3 gF1(DFF / 256, 2048 / 128);               // (16,16)
  const dim3 gF2(DMODEL / 256, 2048 / 128);            // (4,16)

  for (int l = 0; l < N_LAYER; ++l) {
    const size_t bOfs = (size_t)l * DMODEL;
    const size_t oOfs = (size_t)l * DHEAD * (MFEAT / 2);
    const size_t f1b  = (size_t)l * DFF;

    k_gemm_bias<<<gProj, blk, 0, stream>>>(R, WqT + l * MM, bq + bOfs, B1,
                                           BB * SSEQ, DMODEL, DMODEL, 0);
    k_gemm_bias<<<gProj, blk, 0, stream>>>(R, WkT + l * MM, bk + bOfs, B2,
                                           BB * SSEQ, DMODEL, DMODEL, 0);
    k_gemm_bias<<<gProj, blk, 0, stream>>>(R, WvT + l * MM, bv + bOfs, B3,
                                           BB * SSEQ, DMODEL, DMODEL, 0);
    k_phi<<<gAttn, blk, 0, stream>>>(B1, Om + oOfs, QP);
    k_phi<<<gAttn, blk, 0, stream>>>(B2, Om + oOfs, KP);
    k_chunk_kv<<<gAttn, blk, 0, stream>>>(KP, B3, KV, KS);
    k_scan<<<dim3(BB * N_HEAD), blk, 0, stream>>>(KV, KS);
    k_attn<<<gAttn, blk, 0, stream>>>(QP, KP, B3, KV, KS, B1);
    k_gemm_bias<<<gProj, blk, 0, stream>>>(B1, WoT + l * MM, bo + bOfs, B2,
                                           BB * SSEQ, DMODEL, DMODEL, 0);
    k_ln<<<dim3(BB * SSEQ), blk, 0, stream>>>(R, B2, g1 + bOfs, be1 + bOfs, B3);

    for (int mc = 0; mc < 4; ++mc) {   // FFN in 2048-row slices
      const size_t rofs = (size_t)mc * 2048 * DMODEL;
      k_gemm_bias<<<gF1, blk, 0, stream>>>(B3 + rofs, W1T + l * MF, bf1 + f1b, Y1,
                                           2048, DFF, DMODEL, 1);
      k_gemm_bias<<<gF2, blk, 0, stream>>>(Y1, W2T + l * MF, bf2 + bOfs, B2 + rofs,
                                           2048, DMODEL, DFF, 0);
    }
    k_ln<<<dim3(BB * SSEQ), blk, 0, stream>>>(B3, B2, g2 + bOfs, be2 + bOfs, R);
  }
  (void)hipMemcpyAsync(d_out, R, SZ * sizeof(float), hipMemcpyDeviceToDevice, stream);
}
